// PairwiseBlock_21225728377410
// MI455X (gfx1250) — compile-verified, matrix operand order checked by reading, not run on hardware
//
#include <hip/hip_runtime.h>

// ---------------------------------------------------------------------------
// Problem constants (from reference): B=2, N=256, D=128, H=4, DH=32, EXP=4
// ---------------------------------------------------------------------------
#define BB 2
#define NN 256
#define DD 128
#define HH 4
#define DHH 32
#define MM (BB * NN * NN)          // 131072 rows in (b,i,j) space

typedef unsigned short u16;
typedef unsigned int   u32;

#define DEV static __device__ __forceinline__

DEV float bf2f(u16 h) { u32 x = ((u32)h) << 16; float f; __builtin_memcpy(&f, &x, 4); return f; }
DEV u16   f2bf(float f) { u32 x; __builtin_memcpy(&x, &f, 4); u32 r = x + 0x7FFFu + ((x >> 16) & 1u); return (u16)(r >> 16); }
DEV float sigf(float x) { return 1.0f / (1.0f + __expf(-x)); }
DEV float wred_sum(float v) { for (int o = 16; o; o >>= 1) v += __shfl_xor(v, o, 32); return v; }
DEV float wred_max(float v) { for (int o = 16; o; o >>= 1) v = fmaxf(v, __shfl_xor(v, o, 32)); return v; }

// ---------------------------------------------------------------------------
// WMMA fragment types (gfx1250, wave32): v_wmma_f32_16x16x32_bf16
// ---------------------------------------------------------------------------
typedef __bf16 v16bf __attribute__((ext_vector_type(16)));
typedef float  v8f   __attribute__((ext_vector_type(8)));

union FB { v16bf v; u32 d[8]; u16 u[16]; };
union FC { v8f   v; float f[8]; };

// K offset of dword v for this lane-half (ISA 7.12.2, 16-bit A 16x32 layout)
DEV int kbase(int v, int kh) { return (v < 4 ? 2 * v : 16 + 2 * (v - 4)) + 8 * kh; }

// ---------------------------------------------------------------------------
// Generic batched GEMM:  C[z][m,n] = scale * sum_k A[z][m,k] * W[z][n,k]
//   A : bf16, row stride lda, k contiguous
//   W : bf16, element (n,k) at offW + n*sWn + k*sWk
//       STRB=false -> sWk==1, packed dword loads; STRB=true -> strided u16 pair
//   FULL=true  -> Nout % 64 == 0: all 4 n-tiles active, branch-free inner loop
//   C : f32 or bf16, row stride ldc
//   batch offsets: off = (z/zdiv)*hi + (z%zdiv)*lo
//
// Block = 128 threads (4 waves). Block tile 64(M) x 64(N):
//   wave w owns rows [m0+16w, m0+16w+16); computes 4 16x16 n-tiles (4 accs).
// A and B k-slices (64x32 bf16 each) staged in LDS; all fragment reads are
// ds_load_b128, 4 back-to-back v_wmma per wave per K-step sharing one A frag.
// ---------------------------------------------------------------------------
template <bool STRB, bool FULL>
__global__ void k_gemm(const u16* __restrict__ A, const u16* __restrict__ W,
                       void* __restrict__ Cout,
                       int M, int Nout, int K,
                       int lda, int sWn, int sWk, int ldc, int zdiv,
                       long long zAh, long long zAl,
                       long long zWh, long long zWl,
                       long long zCh, long long zCl,
                       float scale, int outBf16) {
  __shared__ u32 As32[64 * 16];          // 64 rows x 32 bf16 (16 dwords/row)
  __shared__ u32 Bs32[64 * 16];          // 64 cols x 32 bf16
  const int tid  = threadIdx.x;
  const int lane = tid & 31;
  const int wave = tid >> 5;
  const int z    = blockIdx.z;
  const long long offA = (long long)(z / zdiv) * zAh + (long long)(z % zdiv) * zAl;
  const long long offW = (long long)(z / zdiv) * zWh + (long long)(z % zdiv) * zWl;
  const long long offC = (long long)(z / zdiv) * zCh + (long long)(z % zdiv) * zCl;
  const int m0  = blockIdx.x * 64;
  const int n0b = blockIdx.y * 64;
  const int mrow = lane & 15;
  const int kh   = lane >> 4;

  FC acc[4];
#pragma unroll
  for (int t = 0; t < 4; ++t)
#pragma unroll
    for (int r = 0; r < 8; ++r) acc[t].f[r] = 0.0f;

  for (int kb = 0; kb < K; kb += 32) {
    __syncthreads();
#pragma unroll
    for (int t = 0; t < 8; ++t) {        // 1024 dwords each, 128 threads
      int idx = tid + t * 128;
      int row = idx >> 4;                // 16 dwords per row
      int dc  = idx & 15;
      int kg  = kb + dc * 2;
      As32[idx] = *(const u32*)(A + offA + (long long)(m0 + row) * lda + kg);
      int n = n0b + row;
      if (!FULL) { if (n > Nout - 1) n = Nout - 1; }     // clamp for Nout<64
      if (STRB) {
        u32 lo = W[offW + (long long)n * sWn + (long long)kg * sWk];
        u32 hi = W[offW + (long long)n * sWn + (long long)(kg + 1) * sWk];
        Bs32[idx] = lo | (hi << 16);
      } else {
        Bs32[idx] = *(const u32*)(W + offW + (long long)n * sWn + kg);
      }
    }
    if (kb + 32 < K)                     // gfx1250 global_prefetch_b8
      __builtin_prefetch(A + offA + (long long)(m0 + (tid & 63)) * lda + kb + 32, 0, 1);
    __syncthreads();

    FB a;
#pragma unroll
    for (int v = 0; v < 8; ++v)
      a.d[v] = As32[(wave * 16 + mrow) * 16 + (kbase(v, kh) >> 1)];
#pragma unroll
    for (int t = 0; t < 4; ++t) {
      if (FULL || (n0b + t * 16 + 16 <= Nout)) {  // block-uniform -> EXEC all-ones
        FB b;
#pragma unroll
        for (int v = 0; v < 8; ++v)
          b.d[v] = Bs32[(t * 16 + (lane & 15)) * 16 + (kbase(v, kh) >> 1)];
        acc[t].v = __builtin_amdgcn_wmma_f32_16x16x32_bf16(false, a.v, false, b.v,
                                                           (short)0, acc[t].v, false, false);
      }
    }
  }

#pragma unroll
  for (int t = 0; t < 4; ++t) {
    if (FULL || (n0b + t * 16 + 16 <= Nout)) {
      int ncol = n0b + t * 16 + (lane & 15);
#pragma unroll
      for (int r = 0; r < 8; ++r) {
        int m = m0 + wave * 16 + r + 8 * kh;
        float val = acc[t].f[r] * scale;
        long long ci = offC + (long long)m * ldc + ncol;
        if (outBf16) ((u16*)Cout)[ci] = f2bf(val);
        else         ((float*)Cout)[ci] = val;
      }
    }
  }
}

// ---------------------------------------------------------------------------
// LayerNorm over last dim (128). One wave per row. transpose permutes output
// row (b,i,j)->(b,j,i). outBf16 selects output dtype.
// ---------------------------------------------------------------------------
__global__ void k_ln(const float* __restrict__ X, const float* __restrict__ g,
                     const float* __restrict__ bta, void* __restrict__ out,
                     int rows, int transpose, int outBf16) {
  int gw = blockIdx.x * 8 + (threadIdx.x >> 5);
  if (gw >= rows) return;
  int lane = threadIdx.x & 31;
  float4 xv = *(const float4*)(X + (long long)gw * 128 + lane * 4);
  float s  = xv.x + xv.y + xv.z + xv.w;
  float s2 = xv.x * xv.x + xv.y * xv.y + xv.z * xv.z + xv.w * xv.w;
  s = wred_sum(s); s2 = wred_sum(s2);
  float mu   = s * (1.0f / 128.0f);
  float var  = s2 * (1.0f / 128.0f) - mu * mu;
  float rstd = rsqrtf(var + 1e-5f);
  int orow = gw;
  if (transpose) { int b = gw >> 16, i = (gw >> 8) & 255, j = gw & 255; orow = (b << 16) | (j << 8) | i; }
  float4 gg = *(const float4*)(g + lane * 4);
  float4 bb = *(const float4*)(bta + lane * 4);
  float o0 = (xv.x - mu) * rstd * gg.x + bb.x;
  float o1 = (xv.y - mu) * rstd * gg.y + bb.y;
  float o2 = (xv.z - mu) * rstd * gg.z + bb.z;
  float o3 = (xv.w - mu) * rstd * gg.w + bb.w;
  if (outBf16) {
    u32 lo = (u32)f2bf(o0) | ((u32)f2bf(o1) << 16);
    u32 hi = (u32)f2bf(o2) | ((u32)f2bf(o3) << 16);
    *(uint2*)((u16*)out + (long long)orow * 128 + lane * 4) = make_uint2(lo, hi);
  } else {
    *(float4*)((float*)out + (long long)orow * 128 + lane * 4) = make_float4(o0, o1, o2, o3);
  }
}

__global__ void k_convert(const float* __restrict__ src, u16* __restrict__ dst, long long n) {
  long long i = (long long)blockIdx.x * blockDim.x + threadIdx.x;
  if (i < n) dst[i] = f2bf(src[i]);
}

// GLU over 512-wide projection + pair mask; out bf16 [M,256] (left|right)
__global__ void k_glu_mask(const float* __restrict__ lr, const unsigned char* __restrict__ mask,
                           u16* __restrict__ out) {
  long long idx = (long long)blockIdx.x * blockDim.x + threadIdx.x;
  int m = (int)(idx >> 8), c = (int)(idx & 255);
  float a  = lr[(long long)m * 512 + c];
  float gg = lr[(long long)m * 512 + 256 + c];
  float o  = a * sigf(gg);
  int b = m >> 16, i = (m >> 8) & 255, j = m & 255;
  if (!(mask[b * 256 + i] && mask[b * 256 + j])) o = 0.0f;
  out[(long long)m * 256 + c] = f2bf(o);
}

// (b,i,k,d) half of GLU output -> (b,d,p,q); swap reads src row (b,q,p)
__global__ void k_transpose_bd(const u16* __restrict__ lrglu, u16* __restrict__ dst,
                               int halfOff, int swap) {
  int idx = blockIdx.x * blockDim.x + threadIdx.x;     // 16,777,216 elements
  int q = idx & 255, p = (idx >> 8) & 255, d = (idx >> 16) & 127, b = idx >> 23;
  int si = swap ? q : p, sj = swap ? p : q;
  int m = (b << 16) | (si << 8) | sj;
  dst[idx] = lrglu[(long long)m * 256 + halfOff + d];
}

// Triangle out: LN over d of O[b,d,i,j] (strided) * sigmoid(og gate) -> bf16
__global__ void k_tri_out_fuse(const float* __restrict__ O, const float* __restrict__ ong,
                               const float* __restrict__ onb, const float* __restrict__ gate,
                               u16* __restrict__ out) {
  int gw = blockIdx.x * 8 + (threadIdx.x >> 5);
  if (gw >= MM) return;
  int lane = threadIdx.x & 31;
  int b = gw >> 16, i = (gw >> 8) & 255, j = gw & 255;
  float x[4];
#pragma unroll
  for (int t = 0; t < 4; ++t) {
    int d = lane * 4 + t;
    x[t] = O[(((long long)(b * 128 + d)) << 16) + (i << 8) + j];
  }
  float s = x[0] + x[1] + x[2] + x[3];
  float s2 = x[0]*x[0] + x[1]*x[1] + x[2]*x[2] + x[3]*x[3];
  s = wred_sum(s); s2 = wred_sum(s2);
  float mu = s * (1.0f / 128.0f);
  float rstd = rsqrtf(s2 * (1.0f / 128.0f) - mu * mu + 1e-5f);
#pragma unroll
  for (int t = 0; t < 4; ++t) {
    int d = lane * 4 + t;
    float o = (x[t] - mu) * rstd * ong[d] + onb[d];
    o *= sigf(gate[(long long)gw * 128 + d]);
    out[(long long)gw * 128 + d] = f2bf(o);
  }
}

// bias[b,h,p,q] = sum_d xn[b,p,q,d] * bw[h,d]   (tiny N=4 GEMM, VALU)
__global__ void k_bias(const u16* __restrict__ xbf, const float* __restrict__ bw,
                       float* __restrict__ biasOut) {
  long long idx = (long long)blockIdx.x * blockDim.x + threadIdx.x;
  int m = (int)(idx >> 2), h = (int)(idx & 3);
  float s = 0.0f;
  for (int d = 0; d < 128; ++d) s += bf2f(xbf[(long long)m * 128 + d]) * bw[h * 128 + d];
  int b = m >> 16, p = (m >> 8) & 255, q = m & 255;
  biasOut[(((long long)(b * 4 + h)) << 16) + (p << 8) + q] = s;
}

// softmax over j (256) with shared bias + mask; logits f32 -> attn bf16
__global__ void k_softmax(const float* __restrict__ logits, const float* __restrict__ bias,
                          const unsigned char* __restrict__ mask, u16* __restrict__ attn) {
  int gw = blockIdx.x * 8 + (threadIdx.x >> 5);     // row = z*256 + i
  int lane = threadIdx.x & 31;
  int z = gw >> 8, i = gw & 255;
  int b = z >> 10, h = z & 3;
  long long lrow = (long long)gw * 256;
  long long brow = (((long long)(b * 4 + h)) << 16) + (i << 8);
  int j0 = lane * 8;
  float v[8];
  float4 l0 = *(const float4*)(logits + lrow + j0);
  float4 l1 = *(const float4*)(logits + lrow + j0 + 4);
  float4 b0 = *(const float4*)(bias + brow + j0);
  float4 b1 = *(const float4*)(bias + brow + j0 + 4);
  v[0]=l0.x+b0.x; v[1]=l0.y+b0.y; v[2]=l0.z+b0.z; v[3]=l0.w+b0.w;
  v[4]=l1.x+b1.x; v[5]=l1.y+b1.y; v[6]=l1.z+b1.z; v[7]=l1.w+b1.w;
#pragma unroll
  for (int e = 0; e < 8; ++e) if (!mask[b * 256 + j0 + e]) v[e] = -1e9f;
  float mx = v[0];
#pragma unroll
  for (int e = 1; e < 8; ++e) mx = fmaxf(mx, v[e]);
  mx = wred_max(mx);
  float sm = 0.0f;
#pragma unroll
  for (int e = 0; e < 8; ++e) { v[e] = __expf(v[e] - mx); sm += v[e]; }
  sm = wred_sum(sm);
  float r = 1.0f / sm;
  u32 w[4];
#pragma unroll
  for (int e = 0; e < 4; ++e)
    w[e] = (u32)f2bf(v[2 * e] * r) | ((u32)f2bf(v[2 * e + 1] * r) << 16);
  *(uint4*)(attn + lrow + j0) = make_uint4(w[0], w[1], w[2], w[3]);
}

__global__ void k_gate_sigmoid(const float* __restrict__ y, const float* __restrict__ g,
                               u16* __restrict__ out, long long n) {
  long long i = (long long)blockIdx.x * blockDim.x + threadIdx.x;
  if (i < n) out[i] = f2bf(y[i] * sigf(g[i]));
}

// SwiGLU: h[M,1024] -> bf16 [M,512] : silu(g)*a, a=h[:, :512], g=h[:,512:]
__global__ void k_swiglu(const float* __restrict__ hbuf, u16* __restrict__ out) {
  long long idx = (long long)blockIdx.x * blockDim.x + threadIdx.x;
  int m = (int)(idx >> 9), c = (int)(idx & 511);
  float a  = hbuf[(long long)m * 1024 + c];
  float gg = hbuf[(long long)m * 1024 + 512 + c];
  out[(long long)m * 512 + c] = f2bf(a * gg * sigf(gg));
}

// x[m] += src[m] (or src[(b,j,i)] when transpose)
__global__ void k_resadd(float* __restrict__ x, const float* __restrict__ src, int transpose) {
  long long idx = (long long)blockIdx.x * blockDim.x + threadIdx.x;
  int m = (int)(idx >> 7), c = (int)(idx & 127);
  int ms = m;
  if (transpose) { int b = m >> 16, i = (m >> 8) & 255, j = m & 255; ms = (b << 16) | (j << 8) | i; }
  x[(long long)m * 128 + c] += src[(long long)ms * 128 + c];
}

// ---------------------------------------------------------------------------
// Host orchestration
// ---------------------------------------------------------------------------
extern "C" void kernel_launch(void* const* d_in, const int* in_sizes, int n_in,
                              void* d_out, int out_size, void* d_ws, size_t ws_size,
                              hipStream_t stream) {
  (void)in_sizes; (void)n_in; (void)out_size; (void)ws_size;
  const float* PR = (const float*)d_in[0];
  const unsigned char* MASK = (const unsigned char*)d_in[1];
  // param flattening order (dict order): ln0..ln4 g/b, tmo, tmi, tas, tae, trans
  const float* LN0G = (const float*)d_in[2];  const float* LN0B = (const float*)d_in[3];
  const float* LN1G = (const float*)d_in[4];  const float* LN1B = (const float*)d_in[5];
  const float* LN2G = (const float*)d_in[6];  const float* LN2B = (const float*)d_in[7];
  const float* LN3G = (const float*)d_in[8];  const float* LN3B = (const float*)d_in[9];
  const float* LN4G = (const float*)d_in[10]; const float* LN4B = (const float*)d_in[11];
  // tmo: norm_g, norm_b, lr_w, og_w, on_g, on_b, out_w   (12..18)
  // tmi: 19..25 ; tas: bias_w,q,k,v,g,o (26..31) ; tae: 32..37 ; trans: w1,w2 (38,39)

  float* X = (float*)d_out;

  char* ws = (char*)d_ws;
  size_t off = 0;
  auto alloc = [&](size_t bytes) -> void* {
    void* p = ws + off;
    off = (off + bytes + 255) & ~(size_t)255;
    return p;
  };

  u16* xbf  = (u16*)alloc((size_t)MM * 128 * 2);
  u16* qbf  = (u16*)alloc((size_t)MM * 128 * 2);
  u16* kbf  = (u16*)alloc((size_t)MM * 128 * 2);
  u16* vbf  = (u16*)alloc((size_t)MM * 128 * 2);
  u16* tgbf = (u16*)alloc((size_t)MM * 128 * 2);
  u16* abf  = (u16*)alloc((size_t)2048 * 65536 * 2);
  float* big0   = (float*)alloc((size_t)MM * 1024 * 4);
  u16*   big1   = (u16*)alloc((size_t)MM * 512 * 2);
  float* gbuf   = (float*)alloc((size_t)MM * 128 * 4);
  float* tmp128 = (float*)alloc((size_t)MM * 128 * 4);
  u16* Lt = (u16*)alloc((size_t)16777216 * 2);
  u16* Rt = (u16*)alloc((size_t)16777216 * 2);
  float* biasbuf = (float*)alloc((size_t)2 * 4 * 65536 * 4);

  auto cvt = [&](const void* srcp, long long n) -> u16* {
    u16* dst = (u16*)alloc((size_t)n * 2);
    k_convert<<<(unsigned)((n + 255) / 256), 256, 0, stream>>>((const float*)srcp, dst, n);
    return dst;
  };
  u16* w_tmo_lr  = cvt(d_in[14], 512 * 128);
  u16* w_tmo_og  = cvt(d_in[15], 128 * 128);
  u16* w_tmo_out = cvt(d_in[18], 128 * 128);
  u16* w_tmi_lr  = cvt(d_in[21], 512 * 128);
  u16* w_tmi_og  = cvt(d_in[22], 128 * 128);
  u16* w_tmi_out = cvt(d_in[25], 128 * 128);
  u16* w_tas_q = cvt(d_in[27], 128 * 128);
  u16* w_tas_k = cvt(d_in[28], 128 * 128);
  u16* w_tas_v = cvt(d_in[29], 128 * 128);
  u16* w_tas_g = cvt(d_in[30], 128 * 128);
  u16* w_tas_o = cvt(d_in[31], 128 * 128);
  u16* w_tae_q = cvt(d_in[33], 128 * 128);
  u16* w_tae_k = cvt(d_in[34], 128 * 128);
  u16* w_tae_v = cvt(d_in[35], 128 * 128);
  u16* w_tae_g = cvt(d_in[36], 128 * 128);
  u16* w_tae_o = cvt(d_in[37], 128 * 128);
  u16* w_w1 = cvt(d_in[38], 1024 * 128);
  u16* w_w2 = cvt(d_in[39], 128 * 512);

  auto gemm = [&](const u16* A, const u16* W, void* C, int M_, int Nout, int K,
                  int lda, int sWn, int sWk, int ldc, int Z, int zdiv,
                  long long zAh, long long zAl, long long zWh, long long zWl,
                  long long zCh, long long zCl, float scale, int outBf16) {
    dim3 g(M_ / 64, (Nout + 63) / 64, Z);
    bool full = (Nout % 64) == 0;
    if (sWk == 1) {
      if (full)
        k_gemm<false, true><<<g, 128, 0, stream>>>(A, W, C, M_, Nout, K, lda, sWn, sWk, ldc,
                                                   zdiv, zAh, zAl, zWh, zWl, zCh, zCl, scale, outBf16);
      else
        k_gemm<false, false><<<g, 128, 0, stream>>>(A, W, C, M_, Nout, K, lda, sWn, sWk, ldc,
                                                    zdiv, zAh, zAl, zWh, zWl, zCh, zCl, scale, outBf16);
    } else {
      if (full)
        k_gemm<true, true><<<g, 128, 0, stream>>>(A, W, C, M_, Nout, K, lda, sWn, sWk, ldc,
                                                  zdiv, zAh, zAl, zWh, zWl, zCh, zCl, scale, outBf16);
      else
        k_gemm<true, false><<<g, 128, 0, stream>>>(A, W, C, M_, Nout, K, lda, sWn, sWk, ldc,
                                                   zdiv, zAh, zAl, zWh, zWl, zCh, zCl, scale, outBf16);
    }
  };

  // x = pairwise_repr (running residual lives in d_out)
  hipMemcpyAsync(X, PR, (size_t)MM * 128 * 4, hipMemcpyDeviceToDevice, stream);

  auto tri_mult = [&](const float* lng, const float* lnb,
                      const float* nmg, const float* nmb,
                      u16* wlr, u16* wog, const float* ong, const float* onb,
                      u16* wout, bool outgoing) {
    k_ln<<<MM / 8, 256, 0, stream>>>(X, lng, lnb, gbuf, MM, 0, 0);     // outer LN -> f32
    k_ln<<<MM / 8, 256, 0, stream>>>(gbuf, nmg, nmb, xbf, MM, 0, 1);   // module LN -> bf16
    gemm(xbf, wlr, big0, MM, 512, 128, 128, 128, 1, 512, 1, 1, 0, 0, 0, 0, 0, 0, 1.0f, 0);
    k_glu_mask<<<(MM * 256) / 256, 256, 0, stream>>>(big0, MASK, big1);
    if (outgoing) {
      k_transpose_bd<<<65536, 256, 0, stream>>>(big1, Lt, 0, 0);    // A' = left[b,i,k,d]
      k_transpose_bd<<<65536, 256, 0, stream>>>(big1, Rt, 128, 0);  // B' = right[b,j,k,d]
    } else {
      k_transpose_bd<<<65536, 256, 0, stream>>>(big1, Lt, 128, 1);  // A' = right[b,k,i,d]
      k_transpose_bd<<<65536, 256, 0, stream>>>(big1, Rt, 0, 1);    // B' = left[b,k,j,d]
    }
    gemm(xbf, wog, gbuf, MM, 128, 128, 128, 128, 1, 128, 1, 1, 0, 0, 0, 0, 0, 0, 1.0f, 0);
    // batched over z=(b,d): O[b,d,i,j] = Lt @ Rt^T  (256x256x256 x 256 batches)
    gemm(Lt, Rt, tmp128, 256, 256, 256, 256, 256, 1, 256, 256, 1,
         65536, 0, 65536, 0, 65536, 0, 1.0f, 0);
    k_tri_out_fuse<<<MM / 8, 256, 0, stream>>>(tmp128, ong, onb, gbuf, tgbf);
    gemm(tgbf, wout, tmp128, MM, 128, 128, 128, 128, 1, 128, 1, 1, 0, 0, 0, 0, 0, 0, 1.0f, 0);
    k_resadd<<<(MM * 128) / 256, 256, 0, stream>>>(X, tmp128, 0);
  };

  auto tri_attn = [&](const float* lng, const float* lnb, const float* bw,
                      u16* wq, u16* wk, u16* wv, u16* wg, u16* wo, bool transpose) {
    k_ln<<<MM / 8, 256, 0, stream>>>(X, lng, lnb, xbf, MM, transpose ? 1 : 0, 1);
    k_bias<<<(MM * 4) / 256, 256, 0, stream>>>(xbf, bw, biasbuf);
    gemm(xbf, wq, qbf, MM, 128, 128, 128, 128, 1, 128, 1, 1, 0, 0, 0, 0, 0, 0, 1.0f, 1);
    gemm(xbf, wk, kbf, MM, 128, 128, 128, 128, 1, 128, 1, 1, 0, 0, 0, 0, 0, 0, 1.0f, 1);
    gemm(xbf, wv, vbf, MM, 128, 128, 128, 128, 1, 128, 1, 1, 0, 0, 0, 0, 0, 0, 1.0f, 1);
    gemm(xbf, wg, gbuf, MM, 128, 128, 128, 128, 1, 128, 1, 1, 0, 0, 0, 0, 0, 0, 1.0f, 0);
    // logits[z=(b,r,h)][i,j] = q·k^T * DH^-0.5 ; K=32 per head
    gemm(qbf, kbf, big0, 256, 256, 32, 128, 128, 1, 256, 2048, 4,
         32768, 32, 32768, 32, 262144, 65536, 0.176776695f, 0);
    k_softmax<<<(2048 * 256) / 8, 256, 0, stream>>>(big0, biasbuf, MASK, abf);
    // o[z][i,d] = attn @ v ; v accessed with k-stride 128 (B operand strided)
    gemm(abf, vbf, tmp128, 256, 32, 256, 256, 1, 128, 128, 2048, 4,
         262144, 65536, 32768, 32, 32768, 32, 1.0f, 0);
    k_gate_sigmoid<<<(MM * 128) / 256, 256, 0, stream>>>(tmp128, gbuf, tgbf, (long long)MM * 128);
    gemm(tgbf, wo, tmp128, MM, 128, 128, 128, 128, 1, 128, 1, 1, 0, 0, 0, 0, 0, 0, 1.0f, 0);
    k_resadd<<<(MM * 128) / 256, 256, 0, stream>>>(X, tmp128, transpose ? 1 : 0);
  };

  // 1) triangle multiplication outgoing / incoming
  tri_mult(LN0G, LN0B, (const float*)d_in[12], (const float*)d_in[13],
           w_tmo_lr, w_tmo_og, (const float*)d_in[16], (const float*)d_in[17], w_tmo_out, true);
  tri_mult(LN1G, LN1B, (const float*)d_in[19], (const float*)d_in[20],
           w_tmi_lr, w_tmi_og, (const float*)d_in[23], (const float*)d_in[24], w_tmi_out, false);

  // 2) triangle attention (starting node / ending node)
  tri_attn(LN2G, LN2B, (const float*)d_in[26], w_tas_q, w_tas_k, w_tas_v, w_tas_g, w_tas_o, false);
  tri_attn(LN3G, LN3B, (const float*)d_in[32], w_tae_q, w_tae_k, w_tae_v, w_tae_g, w_tae_o, true);

  // 3) transition (SwiGLU)
  k_ln<<<MM / 8, 256, 0, stream>>>(X, LN4G, LN4B, xbf, MM, 0, 1);
  gemm(xbf, w_w1, big0, MM, 1024, 128, 128, 128, 1, 1024, 1, 1, 0, 0, 0, 0, 0, 0, 1.0f, 0);
  k_swiglu<<<(MM * 512) / 256, 256, 0, stream>>>(big0, big1);
  gemm(big1, w_w2, tmp128, MM, 128, 512, 512, 512, 1, 128, 1, 1, 0, 0, 0, 0, 0, 0, 1.0f, 0);
  k_resadd<<<(MM * 128) / 256, 256, 0, stream>>>(X, tmp128, 0);
}
